// LlamaSaeModel_68092411511061
// MI455X (gfx1250) — matrix operand order repaired and checked by reading, not compiled, not run on hardware
//
#include <hip/hip_runtime.h>

// ---------------------------------------------------------------------------
// Model constants (match the reference)
// ---------------------------------------------------------------------------
constexpr int B_    = 2;
constexpr int SEQ   = 1024;
constexpr int DIM   = 1024;
constexpr int NH    = 16;
constexpr int HD    = 64;       // DIM / NH
constexpr int HID   = 2816;
constexpr int VOCAB = 32000;
constexpr int MTOK  = B_ * SEQ; // 2048 token rows

typedef __attribute__((ext_vector_type(16))) __bf16        bf16x16;
typedef __attribute__((ext_vector_type(8)))  float         f32x8;
typedef __attribute__((ext_vector_type(4)))  unsigned int  u32x4;
typedef __attribute__((ext_vector_type(8)))  int           i32x8;
typedef __attribute__((ext_vector_type(4)))  int           i32x4;

#if __has_builtin(__builtin_amdgcn_tensor_load_to_lds) && __has_builtin(__builtin_amdgcn_s_wait_tensorcnt)
#define USE_TDM 1
#else
#define USE_TDM 0
#endif

__device__ __forceinline__ f32x8 wmma_bf16(const bf16x16 a, const bf16x16 b, const f32x8 c) {
  // D = A(16x32 bf16) * B(32x16 bf16) + C(16x16 f32)
  return __builtin_amdgcn_wmma_f32_16x16x32_bf16(false, a, false, b, (short)0, c, false, false);
}

__device__ __forceinline__ f32x8 f32x8_zero() {
  f32x8 z = {0.f, 0.f, 0.f, 0.f, 0.f, 0.f, 0.f, 0.f};
  return z;
}

// LDS byte address of a __shared__ object (LDS aperture truncates to low bits).
__device__ __forceinline__ unsigned int lds_addr_of(const void* p) {
  return (unsigned int)(uintptr_t)p;
}

#if USE_TDM
// ---------------------------------------------------------------------------
// Tensor Data Mover: DMA a 2D tile (tile_h rows x tile_w elems, f32) from
// global (row pitch = stride elems) into LDS, packed rows of tile_w.
// Builds D# group0/group1 per CDNA5 ISA; groups 2/3 and the
// trailing descriptor group are unused for a 2D tile (zero-filled).
// This toolchain exposes the 6-arg builtin:
//   (uint32x4 g0, int32x8 g1, int32x4, int32x4, int32x8, i32 cpol)
// ---------------------------------------------------------------------------
__device__ __forceinline__ void tdm_load_2d_f32(unsigned int lds_addr,
                                                const void* gptr,
                                                unsigned int tile_w,
                                                unsigned int tile_h,
                                                unsigned long long stride_elems) {
  const unsigned long long ga = (unsigned long long)(uintptr_t)gptr;
  u32x4 g0;
  g0[0] = 1u;                                           // count=1, user mode
  g0[1] = lds_addr;                                     // LDS byte address
  g0[2] = (unsigned int)(ga & 0xffffffffu);             // global_addr[31:0] at bit 64
  g0[3] = (unsigned int)((ga >> 32) & 0x01ffffffu)      // global_addr[56:32]
        | (2u << 30);                                   // type = 2 ("image")
  i32x8 g1;
  g1[0] = (int)(2u << 16);                              // wg_mask=0, data_size=2 (4B)
  g1[1] = (int)((tile_w & 0xffffu) << 16);              // tensor_dim0[15:0] (== tile_w)
  g1[2] = (int)(((tile_w >> 16) & 0xffffu)              // tensor_dim0[31:16]
        | ((tile_h & 0xffffu) << 16));                  // tensor_dim1[15:0] (== tile_h)
  g1[3] = (int)(((tile_h >> 16) & 0xffffu)              // tensor_dim1[31:16]
        | ((tile_w & 0xffffu) << 16));                  // tile_dim0
  g1[4] = (int)(tile_h & 0xffffu);                      // tile_dim1 (tile_dim2 = 0)
  g1[5] = (int)(unsigned int)(stride_elems & 0xffffffffu);        // dim0_stride[31:0]
  g1[6] = (int)(unsigned int)((stride_elems >> 32) & 0xffffu);    // dim0_stride[47:32]
  g1[7] = 0;                                            // dim1_stride unused (2D)
  i32x4 z4 = {0, 0, 0, 0};
  i32x8 z8 = {0, 0, 0, 0, 0, 0, 0, 0};
  __builtin_amdgcn_tensor_load_to_lds(g0, g1, z4, z4, z8, 0);
}
#endif

// ---------------------------------------------------------------------------
// Embedding gather: h[row,:] = emb[tokens[row],:]
// ---------------------------------------------------------------------------
__global__ __launch_bounds__(256) void embed_kernel(const int* __restrict__ tokens,
                                                    const float* __restrict__ emb,
                                                    float* __restrict__ h) {
  const int row = blockIdx.x;              // 0 .. MTOK-1
  const int tok = tokens[row];
  const float* src = emb + (size_t)tok * DIM;
  float* dst = h + (size_t)row * DIM;
  for (int i = threadIdx.x; i < DIM; i += blockDim.x) dst[i] = src[i];
}

// ---------------------------------------------------------------------------
// RMSNorm: y = x * rsqrt(mean(x^2)+eps) * w    (one row per block, 256 thr)
// ---------------------------------------------------------------------------
__global__ __launch_bounds__(256) void rmsnorm_kernel(const float* __restrict__ x,
                                                      const float* __restrict__ w,
                                                      float* __restrict__ y,
                                                      int dim) {
  const int row = blockIdx.x;
  const int tid = threadIdx.x;
  const float* xr = x + (size_t)row * dim;
  float s = 0.f;
  for (int i = tid; i < dim; i += 256) { float v = xr[i]; s += v * v; }
#pragma unroll
  for (int m = 16; m >= 1; m >>= 1) s += __shfl_xor(s, m, 32);
  __shared__ float red[8];
  if ((tid & 31) == 0) red[tid >> 5] = s;
  __syncthreads();
  float tot = 0.f;
#pragma unroll
  for (int wv = 0; wv < 8; ++wv) tot += red[wv];
  const float inv = rsqrtf(tot / (float)dim + 1e-5f);
  float* yr = y + (size_t)row * dim;
  for (int i = tid; i < dim; i += 256) yr[i] = xr[i] * inv * w[i];
}

// ---------------------------------------------------------------------------
// RoPE (interleaved pairs), applied in place to q and k.
// ---------------------------------------------------------------------------
__global__ __launch_bounds__(256) void rope_kernel(float* __restrict__ q,
                                                   float* __restrict__ k,
                                                   const int* __restrict__ start_pos) {
  const size_t idx = (size_t)blockIdx.x * blockDim.x + threadIdx.x; // MTOK*NH*HD/2
  const int d  = (int)(idx & 31);        // pair index, HD/2 = 32
  const size_t t = idx >> 5;             // (b*SEQ + s)*NH + h
  const int h  = (int)(t & 15);
  const size_t bs = t >> 4;              // b*SEQ + s
  const int s  = (int)(bs & (SEQ - 1));
  // inv_freq = THETA^(-2d/HD); ln(500000) = 13.122363377404328
  const float inv = __expf(-((float)(2 * d) / (float)HD) * 13.122363377404328f);
  const float f = (float)(s + start_pos[0]) * inv;
  float sn, cs;
  __sincosf(f, &sn, &cs);
  const size_t base = bs * DIM + (size_t)h * HD + 2 * d;
  {
    float xe = q[base], xo = q[base + 1];
    q[base]     = xe * cs - xo * sn;
    q[base + 1] = xe * sn + xo * cs;
  }
  {
    float xe = k[base], xo = k[base + 1];
    k[base]     = xe * cs - xo * sn;
    k[base + 1] = xe * sn + xo * cs;
  }
}

// ---------------------------------------------------------------------------
// WMMA GEMM:  C[M,N] = epilogue( A[M,K] @ W[K,N] )
// Tiles: BM=128, BN=64, BK=32; 256 threads = 8 waves, wave grid 4x2,
// each wave computes 32x32 via 2x2 v_wmma_f32_16x16x32_bf16 tiles.
// f32 tiles DMA'd into LDS by the Tensor Data Mover (one issue per block,
// TENSORcnt-tracked), converted f32->bf16 while building WMMA fragments.
// EPI: 0 = none, 1 = += E (residual), 2 = silu, 3 = *= E
// ---------------------------------------------------------------------------
template <int EPI>
__global__ __launch_bounds__(256) void gemm_wmma(const float* __restrict__ A,
                                                 const float* __restrict__ W,
                                                 float* __restrict__ Cout,
                                                 const float* __restrict__ E,
                                                 int M, int N, int K) {
  constexpr int BM = 128, BN = 64, BK = 32;
  __shared__ float Asf[BM * BK];     // [m][k]  (row pitch BK)  16 KB
  __shared__ float Bsf[BK * BN];     // [k][n]  (row pitch BN)   8 KB
  const int tid = threadIdx.x;
  const int m0 = blockIdx.y * BM, n0 = blockIdx.x * BN;
  const int wid = tid >> 5, lane = tid & 31;
  const int wm = wid >> 1, wn = wid & 1;
  const int half = lane >> 4, l16 = lane & 15;

  f32x8 acc[2][2];
#pragma unroll
  for (int i = 0; i < 2; ++i)
#pragma unroll
    for (int j = 0; j < 2; ++j) acc[i][j] = f32x8_zero();

  for (int k0 = 0; k0 < K; k0 += BK) {
#if USE_TDM
    if (tid == 0) {  // TDM ignores EXEC; descriptors are uniform -> one issue per block
      tdm_load_2d_f32(lds_addr_of(Asf), A + (size_t)m0 * K + k0, BK, BM, (unsigned long long)K);
      tdm_load_2d_f32(lds_addr_of(Bsf), W + (size_t)k0 * N + n0, BN, BK, (unsigned long long)N);
      __builtin_amdgcn_s_wait_tensorcnt(0);
    }
#else
    if (k0 + BK < K) {  // global_prefetch_b8 for the next tiles
      __builtin_prefetch(&A[(size_t)(m0 + (tid & 127)) * K + (k0 + BK)], 0, 1);
      __builtin_prefetch(&W[(size_t)(k0 + BK + (tid >> 6)) * N + (n0 + (tid & 63))], 0, 1);
    }
    {  // A tile 128x32 f32, float4 per thread, 4 iters
      const int mrow = tid >> 3;       // 0..31
      const int kq = tid & 7;          // 32-k row in quarters
#pragma unroll
      for (int it = 0; it < 4; ++it) {
        const int mr = mrow + it * 32;
        const float4 vv = *(const float4*)&A[(size_t)(m0 + mr) * K + k0 + kq * 4];
        *(float4*)&Asf[mr * BK + kq * 4] = vv;
      }
    }
    {  // W tile 32x64 f32
      const int kr = tid >> 4;         // 0..15
      const int nq = tid & 15;
#pragma unroll
      for (int it = 0; it < 2; ++it) {
        const int kk2 = kr + it * 16;
        const float4 vv = *(const float4*)&W[(size_t)(k0 + kk2) * N + n0 + nq * 4];
        *(float4*)&Bsf[kk2 * BN + nq * 4] = vv;
      }
    }
#endif
    __syncthreads();

    bf16x16 af[2], bfr[2];
#pragma unroll
    for (int mt = 0; mt < 2; ++mt) {  // A frag: lane half holds K {h*8..h*8+7, 16+h*8..}
      const float* p = &Asf[(wm * 32 + mt * 16 + l16) * BK];
      const int ks = half * 8;
#pragma unroll
      for (int i = 0; i < 8; ++i) {
        af[mt][i]     = (__bf16)p[ks + i];
        af[mt][8 + i] = (__bf16)p[16 + ks + i];
      }
    }
#pragma unroll
    for (int nt = 0; nt < 2; ++nt) {  // B frag: lane half holds K {h*8..h*8+7, 16+h*8..}
      const int col = wn * 32 + nt * 16 + l16;
#pragma unroll
      for (int i = 0; i < 16; ++i) bfr[nt][i] = (__bf16)Bsf[((i < 8 ? 0 : 8) + half * 8 + i) * BN + col];
    }
#pragma unroll
    for (int mt = 0; mt < 2; ++mt)
#pragma unroll
      for (int nt = 0; nt < 2; ++nt)
        acc[mt][nt] = wmma_bf16(af[mt], bfr[nt], acc[mt][nt]);
    __syncthreads();
  }

  // C layout: VGPR r -> M = r + 8*half, N = l16
#pragma unroll
  for (int mt = 0; mt < 2; ++mt)
#pragma unroll
    for (int nt = 0; nt < 2; ++nt)
#pragma unroll
      for (int r = 0; r < 8; ++r) {
        const int row = m0 + wm * 32 + mt * 16 + r + half * 8;
        const int col = n0 + wn * 32 + nt * 16 + l16;
        const size_t idx = (size_t)row * N + col;
        float vv = acc[mt][nt][r];
        if (EPI == 1)      vv += E[idx];
        else if (EPI == 2) vv = vv / (1.f + __expf(-vv));   // silu
        else if (EPI == 3) vv *= E[idx];
        Cout[idx] = vv;
      }
}

// ---------------------------------------------------------------------------
// Flash attention (causal) with WMMA for both q@k^T and P@V.
// Grid: (SEQ/64, NH, B). Block: 128 threads = 4 waves; wave w owns 16 q-rows.
// ---------------------------------------------------------------------------
__global__ __launch_bounds__(128) void attn_kernel(const float* __restrict__ q,
                                                   const float* __restrict__ k,
                                                   const float* __restrict__ v,
                                                   float* __restrict__ o) {
  __shared__ __bf16 Ks[32 * 64];   // [kv n][d]
  __shared__ __bf16 Vs[64 * 32];   // [d][kv n]  (transposed)
  __shared__ float  Sp[4][16 * 32];

  const int tid = threadIdx.x, wid = tid >> 5, lane = tid & 31;
  const int half = lane >> 4, l16 = lane & 15;
  const int b = blockIdx.z, hh = blockIdx.y, qblk = blockIdx.x;
  const int qrow0 = qblk * 64 + wid * 16;
  const float scale = 0.125f;      // 1/sqrt(HD)

  bf16x16 qf[2];
  {
    const int row = qrow0 + l16;
    const float* qp = q + ((size_t)(b * SEQ + row)) * DIM + hh * HD;
#pragma unroll
    for (int kk = 0; kk < 2; ++kk) {
      const int ks = kk * 32 + half * 8;
#pragma unroll
      for (int i = 0; i < 8; ++i) {
        qf[kk][i]     = (__bf16)(qp[ks + i] * scale);
        qf[kk][8 + i] = (__bf16)(qp[ks + 16 + i] * scale);
      }
    }
  }

  f32x8 of[4];
#pragma unroll
  for (int t = 0; t < 4; ++t) of[t] = f32x8_zero();
  float mrow[8], lrow[8];
#pragma unroll
  for (int r = 0; r < 8; ++r) { mrow[r] = -1e30f; lrow[r] = 0.f; }

  const int jend = qblk * 2 + 2;   // causal: cover cols 0 .. qblk*64+63
  for (int j = 0; j < jend; ++j) {
    __syncthreads();
    {  // stage K block (32 x 64), coalesced over d
      const int d = tid & 63;
      const int nlo = tid >> 6;  // 0..1
#pragma unroll
      for (int it = 0; it < 16; ++it) {
        const int nn = nlo + it * 2;
        Ks[nn * 64 + d] = (__bf16)k[((size_t)(b * SEQ + j * 32 + nn)) * DIM + hh * HD + d];
      }
    }
    {  // stage V block transposed -> Vs[d][kv]
      const int d = tid & 63;
      const int nlo = tid >> 6;
#pragma unroll
      for (int it = 0; it < 16; ++it) {
        const int nn = nlo + it * 2;
        Vs[d * 32 + nn] = (__bf16)v[((size_t)(b * SEQ + j * 32 + nn)) * DIM + hh * HD + d];
      }
    }
    __syncthreads();

    // S(16x32) = q(16x64) @ K^T
    f32x8 s[2];
    s[0] = f32x8_zero(); s[1] = f32x8_zero();
#pragma unroll
    for (int kk = 0; kk < 2; ++kk) {
#pragma unroll
      for (int nt = 0; nt < 2; ++nt) {
        bf16x16 bfK;  // B operand: column = kv index, K = d
        const __bf16* p = &Ks[(nt * 16 + l16) * 64 + kk * 32];
        const int ks = half * 8;
#pragma unroll
        for (int i = 0; i < 16; ++i) bfK[i] = p[ks + i + (i < 8 ? 0 : 8)];
        s[nt] = wmma_bf16(qf[kk], bfK, s[nt]);
      }
    }

    // causal mask + online softmax
#pragma unroll
    for (int r = 0; r < 8; ++r) {
      const int gi = qrow0 + r + half * 8;
#pragma unroll
      for (int nt = 0; nt < 2; ++nt) {
        const int gj = j * 32 + nt * 16 + l16;
        if (gj > gi) s[nt][r] = -1e30f;
      }
      float c = fmaxf(s[0][r], s[1][r]);
#pragma unroll
      for (int msk = 8; msk >= 1; msk >>= 1) c = fmaxf(c, __shfl_xor(c, msk, 16));
      const float mn = fmaxf(mrow[r], c);
      const float alpha = __expf(mrow[r] - mn);
      const float p0 = __expf(s[0][r] - mn);
      const float p1 = __expf(s[1][r] - mn);
      s[0][r] = p0; s[1][r] = p1;
      float ls = p0 + p1;
#pragma unroll
      for (int msk = 8; msk >= 1; msk >>= 1) ls += __shfl_xor(ls, msk, 16);
      lrow[r] = lrow[r] * alpha + ls;
      mrow[r] = mn;
#pragma unroll
      for (int t = 0; t < 4; ++t) of[t][r] *= alpha;
    }

    // relayout P (C layout) -> A-fragment layout through LDS
#pragma unroll
    for (int r = 0; r < 8; ++r) {
      const int rr = r + half * 8;
      Sp[wid][rr * 32 + l16]      = s[0][r];
      Sp[wid][rr * 32 + 16 + l16] = s[1][r];
    }
    __syncthreads();
    bf16x16 pf;
    {
      const float* p = &Sp[wid][l16 * 32];
      const int ks = half * 8;
#pragma unroll
      for (int i = 0; i < 8; ++i) { pf[i] = (__bf16)p[ks + i]; pf[8 + i] = (__bf16)p[16 + ks + i]; }
    }
    // O(16x64) += P(16x32) @ V(32x64)
#pragma unroll
    for (int nt = 0; nt < 4; ++nt) {
      bf16x16 vb;  // B operand: column = d, K = kv
      const __bf16* p = &Vs[(nt * 16 + l16) * 32];
      const int ks = half * 8;
#pragma unroll
      for (int i = 0; i < 16; ++i) vb[i] = p[ks + i + (i < 8 ? 0 : 8)];
      of[nt] = wmma_bf16(pf, vb, of[nt]);
    }
  }

#pragma unroll
  for (int nt = 0; nt < 4; ++nt)
#pragma unroll
    for (int r = 0; r < 8; ++r) {
      const int gi = qrow0 + r + half * 8;
      o[((size_t)(b * SEQ + gi)) * DIM + hh * HD + nt * 16 + l16] = of[nt][r] / lrow[r];
    }
}

// ---------------------------------------------------------------------------
// Host-side orchestration
// ---------------------------------------------------------------------------
extern "C" void kernel_launch(void* const* d_in, const int* in_sizes, int n_in,
                              void* d_out, int out_size, void* d_ws, size_t ws_size,
                              hipStream_t stream) {
  (void)in_sizes; (void)n_in; (void)out_size; (void)ws_size;

  const int*   tokens    = (const int*)d_in[0];
  const int*   start_pos = (const int*)d_in[1];
  const float* emb       = (const float*)d_in[2];
  const float* wq        = (const float*)d_in[3];
  const float* wk        = (const float*)d_in[4];
  const float* wv        = (const float*)d_in[5];
  const float* wo        = (const float*)d_in[6];
  const float* w1        = (const float*)d_in[7];
  const float* w2        = (const float*)d_in[8];
  const float* w3        = (const float*)d_in[9];
  const float* attn_nw   = (const float*)d_in[10];
  const float* ffn_nw    = (const float*)d_in[11];
  const float* final_nw  = (const float*)d_in[12];
  const float* w_out     = (const float*)d_in[13];
  float* out = (float*)d_out;

  float* ws = (float*)d_ws;
  const size_t SZ_TOK_DIM = (size_t)MTOK * DIM;   // 2M floats
  const size_t SZ_TOK_HID = (size_t)MTOK * HID;   // 5.77M floats
  float* h  = ws;                    ws += SZ_TOK_DIM;
  float* x  = ws;                    ws += SZ_TOK_DIM;
  float* qb = ws;                    ws += SZ_TOK_DIM;
  float* kb = ws;                    ws += SZ_TOK_DIM;
  float* vb = ws;                    ws += SZ_TOK_DIM;
  float* ob = ws;                    ws += SZ_TOK_DIM;
  float* gb = ws;                    ws += SZ_TOK_HID;
  float* tb = ws;                    ws += SZ_TOK_HID;

  auto gemm = [&](int epi, const float* A, const float* W, float* C, const float* E,
                  int M, int N, int K) {
    dim3 grid(N / 64, M / 128);
    switch (epi) {
      case 0: gemm_wmma<0><<<grid, 256, 0, stream>>>(A, W, C, E, M, N, K); break;
      case 1: gemm_wmma<1><<<grid, 256, 0, stream>>>(A, W, C, E, M, N, K); break;
      case 2: gemm_wmma<2><<<grid, 256, 0, stream>>>(A, W, C, E, M, N, K); break;
      default: gemm_wmma<3><<<grid, 256, 0, stream>>>(A, W, C, E, M, N, K); break;
    }
  };

  embed_kernel<<<MTOK, 256, 0, stream>>>(tokens, emb, h);

  for (int l = 0; l < 4; ++l) {
    const float* Wq = wq + (size_t)l * DIM * DIM;
    const float* Wk = wk + (size_t)l * DIM * DIM;
    const float* Wv = wv + (size_t)l * DIM * DIM;
    const float* Wo = wo + (size_t)l * DIM * DIM;
    const float* W1 = w1 + (size_t)l * DIM * HID;
    const float* W2 = w2 + (size_t)l * HID * DIM;
    const float* W3 = w3 + (size_t)l * DIM * HID;

    rmsnorm_kernel<<<MTOK, 256, 0, stream>>>(h, attn_nw + (size_t)l * DIM, x, DIM);
    gemm(0, x, Wq, qb, nullptr, MTOK, DIM, DIM);
    gemm(0, x, Wk, kb, nullptr, MTOK, DIM, DIM);
    gemm(0, x, Wv, vb, nullptr, MTOK, DIM, DIM);
    rope_kernel<<<(MTOK * NH * (HD / 2)) / 256, 256, 0, stream>>>(qb, kb, start_pos);
    attn_kernel<<<dim3(SEQ / 64, NH, B_), 128, 0, stream>>>(qb, kb, vb, ob);
    gemm(1, ob, Wo, h, h, MTOK, DIM, DIM);                 // h += o @ Wo

    rmsnorm_kernel<<<MTOK, 256, 0, stream>>>(h, ffn_nw + (size_t)l * DIM, x, DIM);
    gemm(2, x, W1, gb, nullptr, MTOK, HID, DIM);           // g = silu(x @ W1)
    gemm(3, x, W3, tb, gb,      MTOK, HID, DIM);           // t = (x @ W3) * g
    gemm(1, tb, W2, h, h,       MTOK, DIM, HID);           // h += t @ W2
  }

  rmsnorm_kernel<<<MTOK, 256, 0, stream>>>(h, final_nw, x, DIM);
  gemm(0, x, w_out, out, nullptr, MTOK, VOCAB, DIM);       // logits
}
